// btnnLayer_68719476736056
// MI455X (gfx1250) — matrix operand order repaired and checked by reading, not compiled, not run on hardware
//
#include <hip/hip_runtime.h>

// ---------------------------------------------------------------------------
// Ternary-weight binary-activation layer for MI455X (gfx1250, wave32, WMMA).
//   Y = (X @ ((pr>0)-(nr>0))^T) > 0
// v3: BK=64, double-buffered LDS, 64x64 register tile per wave (16 WMMA per
// fragment set), async global->LDS staging via GLOBAL_LOAD_ASYNC_TO_LDS_B128
// (ASYNCcnt-tracked, no VGPR staging).
// ---------------------------------------------------------------------------

typedef __attribute__((ext_vector_type(16))) __bf16        v16bf;
typedef __attribute__((ext_vector_type(8)))  float         v8f;
typedef __attribute__((ext_vector_type(4)))  unsigned int  v4u;
typedef __attribute__((ext_vector_type(4)))  int           v4i;

typedef __attribute__((address_space(1))) v4i g_v4i;   // global (device) space
typedef __attribute__((address_space(3))) v4i l_v4i;   // LDS space

union Frag {
    v16bf bf;
    v4u   u[2];
};

#define NTOK 4096
#define DIN  4096
#define DOUT 4096

#define BM 128          // block tile M
#define BN 256          // block tile N
#define BK 64           // staged K depth
#define LDSTR 72        // halves per LDS row (144 B): conflict-free b128 reads

#define A_H (BM * LDSTR)            // 9216 halves
#define B_H (BN * LDSTR)            // 18432 halves
#define BUF_H (A_H + B_H)           // per double-buffer stage
#define SMEM_BYTES (2 * BUF_H * 2)  // 110592 bytes

__device__ __forceinline__ unsigned short f32_to_bf16_rne(float f) {
    unsigned int u = __float_as_uint(f);
    u += 0x7FFFu + ((u >> 16) & 1u);
    return (unsigned short)(u >> 16);
}

// ---- pack X: f32 -> bf16, 8 elements per thread --------------------------
__global__ __launch_bounds__(256) void pack_x_kernel(
    const float* __restrict__ x, unsigned short* __restrict__ xb, int n8)
{
    int i = blockIdx.x * 256 + threadIdx.x;
    if (i >= n8) return;
    const float4* p = (const float4*)x + (size_t)i * 2;
    float4 f0 = p[0], f1 = p[1];
    unsigned int o0 = (unsigned int)f32_to_bf16_rne(f0.x) |
                      ((unsigned int)f32_to_bf16_rne(f0.y) << 16);
    unsigned int o1 = (unsigned int)f32_to_bf16_rne(f0.z) |
                      ((unsigned int)f32_to_bf16_rne(f0.w) << 16);
    unsigned int o2 = (unsigned int)f32_to_bf16_rne(f1.x) |
                      ((unsigned int)f32_to_bf16_rne(f1.y) << 16);
    unsigned int o3 = (unsigned int)f32_to_bf16_rne(f1.z) |
                      ((unsigned int)f32_to_bf16_rne(f1.w) << 16);
    v4u outv = { o0, o1, o2, o3 };
    ((v4u*)xb)[i] = outv;
}

// ---- pack W: ternary (pr>0)-(nr>0) -> exact bf16 {-1,0,1} ----------------
__device__ __forceinline__ unsigned short tern_bf16(float pr, float nr) {
    int s = (pr > 0.0f) - (nr > 0.0f);
    return s == 0 ? (unsigned short)0
                  : (s > 0 ? (unsigned short)0x3F80u : (unsigned short)0xBF80u);
}

__global__ __launch_bounds__(256) void pack_w_kernel(
    const float* __restrict__ pr, const float* __restrict__ nr,
    unsigned short* __restrict__ wb, int n8)
{
    int i = blockIdx.x * 256 + threadIdx.x;
    if (i >= n8) return;
    const float4* pp = (const float4*)pr + (size_t)i * 2;
    const float4* pn = (const float4*)nr + (size_t)i * 2;
    float4 p0 = pp[0], p1 = pp[1];
    float4 q0 = pn[0], q1 = pn[1];
    unsigned int o0 = (unsigned int)tern_bf16(p0.x, q0.x) |
                      ((unsigned int)tern_bf16(p0.y, q0.y) << 16);
    unsigned int o1 = (unsigned int)tern_bf16(p0.z, q0.z) |
                      ((unsigned int)tern_bf16(p0.w, q0.w) << 16);
    unsigned int o2 = (unsigned int)tern_bf16(p1.x, q1.x) |
                      ((unsigned int)tern_bf16(p1.y, q1.y) << 16);
    unsigned int o3 = (unsigned int)tern_bf16(p1.z, q1.z) |
                      ((unsigned int)tern_bf16(p1.w, q1.w) << 16);
    v4u outv = { o0, o1, o2, o3 };
    ((v4u*)wb)[i] = outv;
}

// ---- staging helpers ------------------------------------------------------
__device__ __forceinline__ void copy16_to_lds(const unsigned short* g,
                                              unsigned short* l)
{
    __builtin_amdgcn_global_load_async_to_lds_b128(
        (g_v4i*)g, (l_v4i*)l, 0, 0);
}

__device__ __forceinline__ void wait_stage() {
    __builtin_amdgcn_s_wait_asynccnt(0);
}

// Stage A(128xBK) and B(256xBK) bf16 tiles into LDS. 16B chunks:
// A: 1024 chunks (4/thread), B: 2048 chunks (8/thread).
__device__ __forceinline__ void stage_tiles(
    const unsigned short* __restrict__ xb,
    const unsigned short* __restrict__ wb,
    unsigned short* lA, unsigned short* lB,
    int bm, int bn, int k0, int tid)
{
#pragma unroll
    for (int c = 0; c < 4; ++c) {
        int chunk = tid + c * 256;
        int row = chunk >> 3;            // 0..127
        int col = (chunk & 7) * 8;       // halves 0..56
        copy16_to_lds(xb + (size_t)(bm + row) * DIN + k0 + col,
                      lA + row * LDSTR + col);
    }
#pragma unroll
    for (int c = 0; c < 8; ++c) {
        int chunk = tid + c * 256;
        int row = chunk >> 3;            // 0..255
        int col = (chunk & 7) * 8;
        copy16_to_lds(wb + (size_t)(bn + row) * DIN + k0 + col,
                      lB + row * LDSTR + col);
    }
}

// ---- bf16 WMMA GEMM + threshold ------------------------------------------
// Block: 256 threads = 8 waves (2 M x 4 N), each wave owns 64x64 = 4x4 frags.
__global__ __launch_bounds__(256) void gemm_btnn_kernel(
    const unsigned short* __restrict__ xb,   // [NTOK x DIN] bf16 row-major
    const unsigned short* __restrict__ wb,   // [DOUT x DIN] bf16 row-major
    float* __restrict__ out)                 // [NTOK x DOUT] f32
{
    extern __shared__ __align__(16) unsigned short smem[];

    const int tid  = threadIdx.x;
    const int lane = tid & 31;
    const int wid  = tid >> 5;
    const int wm   = wid & 1;          // M offset wm*64
    const int wn   = wid >> 1;         // N offset wn*64
    const int bm   = blockIdx.y * BM;
    const int bn   = blockIdx.x * BN;

    const int r16 = lane & 15;
    const int hi  = lane >> 4;

    v8f acc[4][4];
#pragma unroll
    for (int i = 0; i < 4; ++i)
#pragma unroll
        for (int j = 0; j < 4; ++j)
#pragma unroll
            for (int r = 0; r < 8; ++r)
                acc[i][j][r] = 0.0f;

    // Prologue: stage tile 0 into buffer 0.
    stage_tiles(xb, wb, smem, smem + A_H, bm, bn, 0, tid);

    const int nkt = DIN / BK;   // 64 tiles
    for (int kt = 0; kt < nkt; ++kt) {
        const int cur = kt & 1;
        unsigned short* lA = smem + cur * BUF_H;
        unsigned short* lB = lA + A_H;

        wait_stage();        // my stage for buf[cur] has landed in LDS
        __syncthreads();     // everyone's has; everyone done reading buf[cur^1]

        if (kt + 1 < nkt) {
            unsigned short* nA = smem + (cur ^ 1) * BUF_H;
            stage_tiles(xb, wb, nA, nA + A_H, bm, bn, (kt + 1) * BK, tid);
        }

#pragma unroll
        for (int ks = 0; ks < BK / 32; ++ks) {
            Frag a[4], b[4];
#pragma unroll
            for (int i = 0; i < 4; ++i) {
                const unsigned short* base =
                    lA + (wm * 64 + i * 16 + r16) * LDSTR + ks * 32;
                a[i].u[0] = *(const v4u*)(base + hi * 8);       // K 0..7 / 8..15
                a[i].u[1] = *(const v4u*)(base + 16 + hi * 8);  // K 16..23/24..31
            }
#pragma unroll
            for (int j = 0; j < 4; ++j) {
                const unsigned short* base =
                    lB + (wn * 64 + j * 16 + r16) * LDSTR + ks * 32;
                b[j].u[0] = *(const v4u*)(base + hi * 8);
                b[j].u[1] = *(const v4u*)(base + 16 + hi * 8);
            }
#pragma unroll
            for (int i = 0; i < 4; ++i)
#pragma unroll
                for (int j = 0; j < 4; ++j)
                    acc[i][j] = __builtin_amdgcn_wmma_f32_16x16x32_bf16(
                        false, a[i].bf, false, b[j].bf,
                        (short)0, acc[i][j], false, false);
        }
    }

    // Threshold + store. C/D layout: VGPR r -> M = r + 8*hi, N = r16.
#pragma unroll
    for (int i = 0; i < 4; ++i) {
#pragma unroll
        for (int j = 0; j < 4; ++j) {
            int row0 = bm + wm * 64 + i * 16 + hi * 8;
            int col  = bn + wn * 64 + j * 16 + r16;
#pragma unroll
            for (int r = 0; r < 8; ++r)
                out[(size_t)(row0 + r) * DOUT + col] =
                    acc[i][j][r] > 0.0f ? 1.0f : 0.0f;
        }
    }
}

// ---------------------------------------------------------------------------
extern "C" void kernel_launch(void* const* d_in, const int* in_sizes, int n_in,
                              void* d_out, int out_size, void* d_ws, size_t ws_size,
                              hipStream_t stream)
{
    const float* x  = (const float*)d_in[0];
    const float* pr = (const float*)d_in[1];
    const float* nr = (const float*)d_in[2];
    float*       out = (float*)d_out;

    unsigned short* xb = (unsigned short*)d_ws;                  // 32 MB
    unsigned short* wb = xb + (size_t)DOUT * DIN;                // 32 MB

    const int nx8 = (NTOK * DIN) / 8;
    const int nw8 = (DOUT * DIN) / 8;

    pack_x_kernel<<<(nx8 + 255) / 256, 256, 0, stream>>>(x, xb, nx8);
    pack_w_kernel<<<(nw8 + 255) / 256, 256, 0, stream>>>(pr, nr, wb, nw8);

    (void)hipFuncSetAttribute((const void*)gemm_btnn_kernel,
                              hipFuncAttributeMaxDynamicSharedMemorySize,
                              SMEM_BYTES);

    dim3 grid(DOUT / BN, NTOK / BM);   // 16 x 32 blocks
    gemm_btnn_kernel<<<grid, 256, SMEM_BYTES, stream>>>(xb, wb, out);
}
